// DGCNN_Generator_36575941492862
// MI455X (gfx1250) — matrix operand order repaired and checked by reading, not compiled, not run on hardware
//
#include <hip/hip_runtime.h>
#include <hip/hip_bf16.h>

// ---------------------------------------------------------------------------
// DGCNN generator for MI455X (gfx1250, wave32, WMMA f32_16x16x32_f16)
// ---------------------------------------------------------------------------

typedef __attribute__((ext_vector_type(16))) _Float16 v16h;
typedef __attribute__((ext_vector_type(8)))  float    v8f;

#define KNN 10

// ---------------------------------------------------------------------------
// prep: f32 features -> f16 padded [pt][CPAD]
// ---------------------------------------------------------------------------
__global__ __launch_bounds__(256)
void pad_f16_kernel(const float* __restrict__ x, _Float16* __restrict__ xh,
                    int CRAW, int CPAD, int total) {
  int tid = blockIdx.x * 256 + threadIdx.x;
  if (tid >= total) return;
  int c  = tid % CPAD;
  int pt = tid / CPAD;
  xh[tid] = (c < CRAW) ? (_Float16)x[(size_t)pt * CRAW + c] : (_Float16)0.0f;
}

__global__ __launch_bounds__(256)
void sqnorm_kernel(const float* __restrict__ x, float* __restrict__ sq,
                   int CRAW, int npts) {
  int pt = blockIdx.x * 256 + threadIdx.x;
  if (pt >= npts) return;
  float s = 0.0f;
  for (int c = 0; c < CRAW; ++c) {
    float v = x[(size_t)pt * CRAW + c];
    s += v * v;
  }
  sq[pt] = s;
}

// ---------------------------------------------------------------------------
// prep: weights f32 [KRAW][NOUT] -> f16 swizzled into WMMA B-fragment order.
// frag(kc, nt) occupies 512 contiguous halves:
//   out[(((kc*(NOUT/16)+nt)*32)+lane)*16+t] = W[kc*32+(lane>>4)*16+t][nt*16+(lane&15)]
// Rows >= KRAW are zero-padded.
// ---------------------------------------------------------------------------
__global__ __launch_bounds__(256)
void prep_w_kernel(const float* __restrict__ w, _Float16* __restrict__ out,
                   int KRAW, int KPAD, int NOUT) {
  int tid = blockIdx.x * 256 + threadIdx.x;
  int total = KPAD * NOUT;
  if (tid >= total) return;
  int t    = tid & 15;
  int lane = (tid >> 4) & 31;
  int rest = tid >> 9;
  int ntpg = NOUT >> 4;
  int nt = rest % ntpg;
  int kc = rest / ntpg;
  int hi = lane >> 4, nl = lane & 15;
  int k = kc * 32 + hi * 16 + t;
  int n = nt * 16 + nl;
  float v = (k < KRAW) ? w[(size_t)k * NOUT + n] : 0.0f;
  out[tid] = (_Float16)v;
}

// ---------------------------------------------------------------------------
// Fused pairwise-distance + top-K kernel.
// Block = 128 threads = 4 waves per 16-row tile. Each wave handles 1/4 of the
// column tiles with a software-pipelined (double-buffered) B-fragment load,
// keeps a register-resident sorted top-10, and the partial lists are merged
// through LDS at the end. Self is included (dist(i,i)=~0 is the minimum),
// matching top_k(-dist).
// ---------------------------------------------------------------------------
template<int CPAD>
__global__ __launch_bounds__(128)
void knn_kernel(const _Float16* __restrict__ xh, const float* __restrict__ sq,
                int* __restrict__ idx, int N) {
  __shared__ float lds_d[4][16][17];
  __shared__ float cand_d[16][4 * KNN];
  __shared__ int   cand_j[16][4 * KNN];

  const int w    = threadIdx.x >> 5;
  const int lane = threadIdx.x & 31;
  const int m  = lane & 15;
  const int hi = lane >> 4;
  const int ntile = N >> 4;
  const int b  = blockIdx.x / ntile;
  const int it = blockIdx.x % ntile;
  const int base = b * N;

  // A fragments: 16x32 f16 per chunk, contiguous k = 16*hi + s (lane m holds row m)
  v16h A[CPAD / 32];
  {
    const _Float16* p = xh + (size_t)(base + it * 16 + m) * CPAD;
#pragma unroll
    for (int kc = 0; kc < CPAD / 32; ++kc) {
#pragma unroll
      for (int t = 0; t < 8; ++t) {
        A[kc][t]     = p[kc * 32 + hi * 16 + t];
        A[kc][8 + t] = p[kc * 32 + hi * 16 + 8 + t];
      }
    }
  }
  float sqi[8];
#pragma unroll
  for (int v = 0; v < 8; ++v) sqi[v] = sq[base + it * 16 + v + 8 * hi];

  float bd[KNN];
  int   bj[KNN];
#pragma unroll
  for (int t = 0; t < KNN; ++t) { bd[t] = 3.0e38f; bj[t] = 0; }

  const int span = ntile >> 2;         // column tiles per wave (128/4 = 32)
  const int jbeg = w * span;
  const int jend = jbeg + span;

  // B[k][n] = X[col_n][k]; lane n holds 16 contiguous k-values
  auto loadB = [&](int jt, v16h (&B)[CPAD / 32], float& sqj) {
    const int col = jt * 16 + m;
    sqj = sq[base + col];
#pragma unroll
    for (int kc = 0; kc < CPAD / 32; ++kc) {
      const _Float16* q = xh + (size_t)(base + col) * CPAD + kc * 32 + hi * 16;
#pragma unroll
      for (int t = 0; t < 16; ++t) B[kc][t] = q[t];
    }
  };

  auto process = [&](int jt, v16h (&B)[CPAD / 32], float sqj) {
    v8f acc = {};
#pragma unroll
    for (int kc = 0; kc < CPAD / 32; ++kc)
      acc = __builtin_amdgcn_wmma_f32_16x16x32_f16(
          false, A[kc], false, B[kc], (short)0, acc, false, false);
#pragma unroll
    for (int v = 0; v < 8; ++v)
      lds_d[w][v + 8 * hi][m] = sqi[v] + sqj - 2.0f * acc[v];
    // in-wave LDS ordering guarantees the loads below see the stores above
    if (hi == 0) {
#pragma unroll 1
      for (int n = 0; n < 16; ++n) {
        float d = lds_d[w][m][n];
        int   j = jt * 16 + n;
        if (d < bd[KNN - 1]) {
          bd[KNN - 1] = d; bj[KNN - 1] = j;
#pragma unroll
          for (int t = KNN - 1; t >= 1; --t) {
            if (bd[t] < bd[t - 1]) {
              float td = bd[t]; bd[t] = bd[t - 1]; bd[t - 1] = td;
              int   tj = bj[t]; bj[t] = bj[t - 1]; bj[t - 1] = tj;
            }
          }
        }
      }
    }
  };

  // software pipeline: two named register buffers (compile-time indices)
  v16h B0[CPAD / 32], B1[CPAD / 32];
  float sqj0, sqj1;
  loadB(jbeg, B0, sqj0);
#pragma unroll 1
  for (int jt = jbeg; jt < jend; jt += 2) {
    loadB(jt + 1, B1, sqj1);
    process(jt, B0, sqj0);
    if (jt + 2 < jend) loadB(jt + 2, B0, sqj0);
    process(jt + 1, B1, sqj1);
  }

  // publish per-wave candidates, then wave 0 merges 4 sorted lists of 10
  if (hi == 0) {
#pragma unroll
    for (int t = 0; t < KNN; ++t) {
      cand_d[m][w * KNN + t] = bd[t];
      cand_j[m][w * KNN + t] = bj[t];
    }
  }
  __syncthreads();
  if (threadIdx.x < 16) {
    const int r = threadIdx.x;
    float fd[KNN];
    int   fj[KNN];
#pragma unroll
    for (int t = 0; t < KNN; ++t) { fd[t] = 3.0e38f; fj[t] = 0; }
#pragma unroll 1
    for (int s = 0; s < 4 * KNN; ++s) {
      float d = cand_d[r][s];
      int   j = cand_j[r][s];
      if (d < fd[KNN - 1]) {
        fd[KNN - 1] = d; fj[KNN - 1] = j;
#pragma unroll
        for (int t = KNN - 1; t >= 1; --t) {
          if (fd[t] < fd[t - 1]) {
            float td = fd[t]; fd[t] = fd[t - 1]; fd[t - 1] = td;
            int   tj = fj[t]; fj[t] = fj[t - 1]; fj[t - 1] = tj;
          }
        }
      }
    }
#pragma unroll
    for (int t = 0; t < KNN; ++t)
      idx[(size_t)(base + it * 16 + r) * KNN + t] = fj[t];
  }
}

// ---------------------------------------------------------------------------
// EdgeConv MLP: one wave per point. The point's K=10 neighbors form a 16-row
// M-tile (rows 10..15 duplicate neighbor 0 so max-over-k is unaffected).
//   h1 = relu(feat @ W1 + b1); y = max_k(h1 @ W2 + b2)
// W1/W2 are pre-swizzled f16 fragments (contiguous 32B per lane per fragment).
// Block = 128 threads = 4 waves, each wave an independent point (per-wave LDS
// slabs; __syncthreads trip counts match across waves).
// ---------------------------------------------------------------------------
template<int CRAW, int CIN, int C1, int C2>
__global__ __launch_bounds__(128)
void edge_mlp_kernel(const float* __restrict__ x, const int* __restrict__ idx,
                     const _Float16* __restrict__ w1s, const float* __restrict__ b1,
                     const _Float16* __restrict__ w2s, const float* __restrict__ b2,
                     float* __restrict__ y, int N) {
  __shared__ _Float16 feat_s[4][16 * CIN];
  __shared__ _Float16 h1_s[4][16 * C1];
  __shared__ int      nb_s[4][16];
  const int w    = threadIdx.x >> 5;
  const int lane = threadIdx.x & 31;
  const int m  = lane & 15;
  const int hi = lane >> 4;
  const int pg = blockIdx.x * 4 + w;
  const int b  = pg / N;
  const int i  = pg % N;
  const size_t pbase = (size_t)(b * N + i);

  // stage neighbor indices once per wave (rows 10..15 duplicate neighbor 0)
  if (hi == 0)
    nb_s[w][m] = idx[pbase * KNN + (m < KNN ? m : 0)];
  // in-wave LDS ordering: subsequent loads by this wave see the store

  // ---- build feat tile (16 rows x CIN) in LDS, f16 ----
  const float* xi = x + pbase * CRAW;
  for (int e = lane; e < 16 * CIN; e += 32) {
    int kk = e / CIN;
    int c  = e % CIN;
    float v = 0.0f;
    if (c < CRAW) {
      v = xi[c];
    } else if (c < 2 * CRAW) {
      int cc = c - CRAW;
      int j  = nb_s[w][kk];
      v = x[(size_t)(b * N + j) * CRAW + cc] - xi[cc];
    }
    feat_s[w][e] = (_Float16)v;
  }
  __syncthreads();

  // ---- layer 1: h1 = relu(feat @ W1 + b1) ----
  v16h A1[CIN / 32];
#pragma unroll
  for (int kc = 0; kc < CIN / 32; ++kc) {
#pragma unroll
    for (int t = 0; t < 8; ++t) {
      A1[kc][t]     = feat_s[w][m * CIN + kc * 32 + hi * 16 + t];
      A1[kc][8 + t] = feat_s[w][m * CIN + kc * 32 + hi * 16 + 8 + t];
    }
  }
#pragma unroll
  for (int nt = 0; nt < C1 / 16; ++nt) {
    float bv = b1[nt * 16 + m];
    v8f acc;
#pragma unroll
    for (int v = 0; v < 8; ++v) acc[v] = bv;
#pragma unroll
    for (int kc = 0; kc < CIN / 32; ++kc) {
      v16h Bf;
      const _Float16* q = w1s + (size_t)((kc * (C1 / 16) + nt) * 32 + lane) * 16;
#pragma unroll
      for (int t = 0; t < 16; ++t) Bf[t] = q[t];
      acc = __builtin_amdgcn_wmma_f32_16x16x32_f16(
          false, A1[kc], false, Bf, (short)0, acc, false, false);
    }
#pragma unroll
    for (int v = 0; v < 8; ++v) {
      float r = fmaxf(acc[v], 0.0f);
      h1_s[w][(v + 8 * hi) * C1 + nt * 16 + m] = (_Float16)r;
    }
  }
  __syncthreads();

  // ---- layer 2: y = max_k(h1 @ W2 + b2) ----
  v16h A2[C1 / 32];
#pragma unroll
  for (int kc = 0; kc < C1 / 32; ++kc) {
#pragma unroll
    for (int t = 0; t < 8; ++t) {
      A2[kc][t]     = h1_s[w][m * C1 + kc * 32 + hi * 16 + t];
      A2[kc][8 + t] = h1_s[w][m * C1 + kc * 32 + hi * 16 + 8 + t];
    }
  }
#pragma unroll
  for (int nt = 0; nt < C2 / 16; ++nt) {
    float bv = b2[nt * 16 + m];
    v8f acc;
#pragma unroll
    for (int v = 0; v < 8; ++v) acc[v] = bv;
#pragma unroll
    for (int kc = 0; kc < C1 / 32; ++kc) {
      v16h Bf;
      const _Float16* q = w2s + (size_t)((kc * (C2 / 16) + nt) * 32 + lane) * 16;
#pragma unroll
      for (int t = 0; t < 16; ++t) Bf[t] = q[t];
      acc = __builtin_amdgcn_wmma_f32_16x16x32_f16(
          false, A2[kc], false, Bf, (short)0, acc, false, false);
    }
    // max over the 16 k-rows: 8 rows per lane half, then swap halves
    float mx = acc[0];
#pragma unroll
    for (int v = 1; v < 8; ++v) mx = fmaxf(mx, acc[v]);
    float other = __shfl_xor(mx, 16, 32);
    mx = fmaxf(mx, other);
    if (hi == 0)
      y[pbase * C2 + nt * 16 + m] = mx;
  }
}

// ---------------------------------------------------------------------------
// global max over N of concat([x1,x2,x3,x4]) -> two-stage reduction
// ---------------------------------------------------------------------------
__global__ __launch_bounds__(320)
void gmax_part_kernel(const float* __restrict__ x1, const float* __restrict__ x2,
                      const float* __restrict__ x3, const float* __restrict__ x4,
                      float* __restrict__ gpart, int N, int S) {
  int b = blockIdx.x;
  int s = blockIdx.y;
  int c = threadIdx.x;   // 0..319
  const float* src;
  int C, off;
  if (c < 64)       { src = x1; C = 64;  off = c; }
  else if (c < 128) { src = x2; C = 64;  off = c - 64; }
  else if (c < 192) { src = x3; C = 64;  off = c - 128; }
  else              { src = x4; C = 128; off = c - 192; }
  int n0 = s * (N / S), n1 = n0 + N / S;
  float mx = -3.0e38f;
  for (int n = n0; n < n1; ++n)
    mx = fmaxf(mx, src[(size_t)(b * N + n) * C + off]);
  gpart[((size_t)b * S + s) * 320 + c] = mx;
}

__global__ __launch_bounds__(320)
void gmax_final_kernel(const float* __restrict__ gpart, float* __restrict__ g, int S) {
  int b = blockIdx.x;
  int c = threadIdx.x;
  float mx = -3.0e38f;
  for (int s = 0; s < S; ++s)
    mx = fmaxf(mx, gpart[((size_t)b * S + s) * 320 + c]);
  g[b * 384 + c] = mx;
}

// ---------------------------------------------------------------------------
// tooth embedding -> g[b][320..383]
// ---------------------------------------------------------------------------
__global__ __launch_bounds__(64)
void emb_kernel(const int* __restrict__ tooth, const float* __restrict__ table,
                const float* __restrict__ w, const float* __restrict__ bias,
                float* __restrict__ g) {
  int b = blockIdx.x;
  int o = threadIdx.x;   // 0..63
  int t = tooth[b];
  float acc = bias[o];
  for (int c = 0; c < 64; ++c)
    acc += table[t * 64 + c] * w[c * 64 + o];
  g[b * 384 + 320 + o] = acc;
}

// ---------------------------------------------------------------------------
// dense head: out[b][o] = (relu?)(in[b] . W[:,o] + bias[o])
// ---------------------------------------------------------------------------
__global__ __launch_bounds__(128)
void fc_kernel(const float* __restrict__ in, const float* __restrict__ w,
               const float* __restrict__ bias, float* __restrict__ out,
               int IN, int OUT, int relu) {
  int b = blockIdx.x;
  int o = blockIdx.y * blockDim.x + threadIdx.x;
  if (o >= OUT) return;
  float acc = bias[o];
  const float* g = in + (size_t)b * IN;
  for (int c = 0; c < IN; ++c)
    acc += g[c] * w[(size_t)c * OUT + o];
  if (relu) acc = fmaxf(acc, 0.0f);
  out[(size_t)b * OUT + o] = acc;
}

// ---------------------------------------------------------------------------
// orchestration
// ---------------------------------------------------------------------------
extern "C" void kernel_launch(void* const* d_in, const int* in_sizes, int n_in,
                              void* d_out, int out_size, void* d_ws, size_t ws_size,
                              hipStream_t stream) {
  (void)in_sizes; (void)n_in; (void)out_size; (void)ws_size;

  const int B = 8, N = 2048;
  const int NP = B * N;
  const int GS = 16;   // gmax slices

  const float* pos        = (const float*)d_in[0];
  const int*   tooth_n    = (const int*)  d_in[1];
  const float* emb_table  = (const float*)d_in[2];
  const float* conv_emb_w = (const float*)d_in[3];
  const float* conv_emb_b = (const float*)d_in[4];
  const float* c1_w1 = (const float*)d_in[5];
  const float* c1_b1 = (const float*)d_in[6];
  const float* c1_w2 = (const float*)d_in[7];
  const float* c1_b2 = (const float*)d_in[8];
  const float* c2_w1 = (const float*)d_in[9];
  const float* c2_b1 = (const float*)d_in[10];
  const float* c2_w2 = (const float*)d_in[11];
  const float* c2_b2 = (const float*)d_in[12];
  const float* c3_w1 = (const float*)d_in[13];
  const float* c3_b1 = (const float*)d_in[14];
  const float* c3_w2 = (const float*)d_in[15];
  const float* c3_b2 = (const float*)d_in[16];
  const float* c4_w1 = (const float*)d_in[17];
  const float* c4_b1 = (const float*)d_in[18];
  const float* c4_w2 = (const float*)d_in[19];
  const float* c4_b2 = (const float*)d_in[20];
  const float* enc_w  = (const float*)d_in[21];
  const float* enc_b  = (const float*)d_in[22];
  const float* dec_w1 = (const float*)d_in[23];
  const float* dec_b1 = (const float*)d_in[24];
  const float* dec_w2 = (const float*)d_in[25];
  const float* dec_b2 = (const float*)d_in[26];

  // workspace carve-up
  char*  ws  = (char*)d_ws;
  size_t off = 0;
  auto carve = [&](size_t bytes) -> void* {
    void* p = ws + off;
    off = (off + bytes + 255) & ~(size_t)255;
    return p;
  };
  _Float16* xh  = (_Float16*)carve((size_t)NP * 64 * 2);
  float*    sq  = (float*)   carve((size_t)NP * 4);
  int*      idx = (int*)     carve((size_t)NP * KNN * 4);
  float*    x1  = (float*)   carve((size_t)NP * 64 * 4);
  float*    x2  = (float*)   carve((size_t)NP * 64 * 4);
  float*    x3  = (float*)   carve((size_t)NP * 64 * 4);
  float*    x4  = (float*)   carve((size_t)NP * 128 * 4);
  _Float16* w1s = (_Float16*)carve((size_t)128 * 128 * 2);
  _Float16* w2s = (_Float16*)carve((size_t)128 * 128 * 2);
  float*    gpart = (float*) carve((size_t)B * GS * 320 * 4);
  float*    g     = (float*) carve((size_t)B * 384 * 4);
  float*    hbuf  = (float*) carve((size_t)B * 512 * 4);
  float*    tbuf  = (float*) carve((size_t)B * 1024 * 4);

  const int knn_grid = B * (N / 16);
  const int mlp_grid = NP / 4;

  // ---------------- EdgeConv 1 (pos, CRAW=3 -> 64) ----------------
  pad_f16_kernel<<<(NP * 32 + 255) / 256, 256, 0, stream>>>(pos, xh, 3, 32, NP * 32);
  sqnorm_kernel<<<(NP + 255) / 256, 256, 0, stream>>>(pos, sq, 3, NP);
  knn_kernel<32><<<knn_grid, 128, 0, stream>>>(xh, sq, idx, N);
  prep_w_kernel<<<(32 * 64 + 255) / 256, 256, 0, stream>>>(c1_w1, w1s, 6, 32, 64);
  prep_w_kernel<<<(64 * 64 + 255) / 256, 256, 0, stream>>>(c1_w2, w2s, 64, 64, 64);
  edge_mlp_kernel<3, 32, 64, 64><<<mlp_grid, 128, 0, stream>>>(
      pos, idx, w1s, c1_b1, w2s, c1_b2, x1, N);

  // ---------------- EdgeConv 2 (x1, 64 -> 64) ----------------
  pad_f16_kernel<<<(NP * 64 + 255) / 256, 256, 0, stream>>>(x1, xh, 64, 64, NP * 64);
  sqnorm_kernel<<<(NP + 255) / 256, 256, 0, stream>>>(x1, sq, 64, NP);
  knn_kernel<64><<<knn_grid, 128, 0, stream>>>(xh, sq, idx, N);
  prep_w_kernel<<<(128 * 64 + 255) / 256, 256, 0, stream>>>(c2_w1, w1s, 128, 128, 64);
  prep_w_kernel<<<(64 * 64 + 255) / 256, 256, 0, stream>>>(c2_w2, w2s, 64, 64, 64);
  edge_mlp_kernel<64, 128, 64, 64><<<mlp_grid, 128, 0, stream>>>(
      x1, idx, w1s, c2_b1, w2s, c2_b2, x2, N);

  // ---------------- EdgeConv 3 (x2, 64 -> 64) ----------------
  pad_f16_kernel<<<(NP * 64 + 255) / 256, 256, 0, stream>>>(x2, xh, 64, 64, NP * 64);
  sqnorm_kernel<<<(NP + 255) / 256, 256, 0, stream>>>(x2, sq, 64, NP);
  knn_kernel<64><<<knn_grid, 128, 0, stream>>>(xh, sq, idx, N);
  prep_w_kernel<<<(128 * 64 + 255) / 256, 256, 0, stream>>>(c3_w1, w1s, 128, 128, 64);
  prep_w_kernel<<<(64 * 64 + 255) / 256, 256, 0, stream>>>(c3_w2, w2s, 64, 64, 64);
  edge_mlp_kernel<64, 128, 64, 64><<<mlp_grid, 128, 0, stream>>>(
      x2, idx, w1s, c3_b1, w2s, c3_b2, x3, N);

  // ---------------- EdgeConv 4 (x3, 64 -> 128) ----------------
  pad_f16_kernel<<<(NP * 64 + 255) / 256, 256, 0, stream>>>(x3, xh, 64, 64, NP * 64);
  sqnorm_kernel<<<(NP + 255) / 256, 256, 0, stream>>>(x3, sq, 64, NP);
  knn_kernel<64><<<knn_grid, 128, 0, stream>>>(xh, sq, idx, N);
  prep_w_kernel<<<(128 * 128 + 255) / 256, 256, 0, stream>>>(c4_w1, w1s, 128, 128, 128);
  prep_w_kernel<<<(128 * 128 + 255) / 256, 256, 0, stream>>>(c4_w2, w2s, 128, 128, 128);
  edge_mlp_kernel<64, 128, 128, 128><<<mlp_grid, 128, 0, stream>>>(
      x3, idx, w1s, c4_b1, w2s, c4_b2, x4, N);

  // ---------------- global max + embedding + head ----------------
  gmax_part_kernel<<<dim3(B, GS), 320, 0, stream>>>(x1, x2, x3, x4, gpart, N, GS);
  gmax_final_kernel<<<B, 320, 0, stream>>>(gpart, g, GS);
  emb_kernel<<<B, 64, 0, stream>>>(tooth_n, emb_table, conv_emb_w, conv_emb_b, g);
  fc_kernel<<<dim3(B, 4), 128, 0, stream>>>(g, enc_w, enc_b, hbuf, 384, 512, 1);
  fc_kernel<<<dim3(B, 8), 128, 0, stream>>>(hbuf, dec_w1, dec_b1, tbuf, 512, 1024, 1);
  fc_kernel<<<dim3(B, 24), 128, 0, stream>>>(tbuf, dec_w2, dec_b2, (float*)d_out,
                                             1024, 3072, 0);
}
